// DynamicReductionNetwork_18846316495272
// MI455X (gfx1250) — matrix operand order repaired and checked by reading, not compiled, not run on hardware
//
#include <hip/hip_runtime.h>
#include <hip/hip_bf16.h>

// ---------------------------------------------------------------------------
// CDNA5 (gfx1250, wave32) implementation of the Dynamic Reduction Network.
// All dense layers (input MLP, EdgeConv MLP) and the knn Gram matrix run on
// V_WMMA_F32_16X16X32_F16 (f16 in / f32 accumulate). Sequential pieces
// (bitonic sort, greedy matching) are VALU/LDS kernels.
// ---------------------------------------------------------------------------

typedef __attribute__((ext_vector_type(16))) _Float16 v16h;
typedef __attribute__((ext_vector_type(8)))  float    v8f;

#define G_      4                 // graphs (B)
#define NP_     2048              // points per graph
#define INDIM_  5
#define HID_    128
#define KNN_    8
#define EMAX_   (NP_*KNN_)        // 16384 max edges per graph
#define SORTN_  16384             // power of two >= EMAX_
#define MAXLVL_ 32                // >= pooling levels needed (n roughly halves)

__device__ __forceinline__ int dmin_(int a, int b) { return a < b ? a : b; }

// ---------------------------------------------------------------------------
// Generic WMMA GEMM:  C[M x N] = act( A[M x K] * W[N x K]^T + bias )
// One wave per 16x16 output tile. mMode: 0 -> M = mFixed (single "graph"),
// 1 -> M = n_g * k_eff (edge rows), 2 -> M = n_g.  Rows >= M are stored as
// zero into the f16 output (keeps downstream K-padding clean).
// ---------------------------------------------------------------------------
__global__ __launch_bounds__(32)
void gemm_wmma(const _Float16* __restrict__ A, long aStride,
               const _Float16* __restrict__ W,
               const float* __restrict__ bias,
               float* __restrict__ Co, long coStride,
               _Float16* __restrict__ Ch, long chStride,
               const int* __restrict__ nCur, int mMode, int mFixed,
               int N, int Kd, int act)
{
  int g = blockIdx.z;
  int Mg;
  if (mMode == 0) {
    Mg = mFixed;
  } else {
    int n = nCur[g];
    if (n <= 1) return;
    Mg = (mMode == 1) ? n * dmin_(KNN_, n - 1) : n;
  }
  int r0 = blockIdx.y << 4;
  if (r0 >= Mg) return;
  int n0 = blockIdx.x << 4;
  int l  = threadIdx.x;
  int lm = l & 15, lg = l >> 4;

  const _Float16* Ag = A + (long)g * aStride;
  v8f acc;
  float bv = bias ? bias[n0 + lm] : 0.0f;
  #pragma unroll
  for (int r = 0; r < 8; ++r) acc[r] = bv;

  // A layout (16-bit A 16x32): lanes 0-15 -> M=lane, halves j<8 -> K=j,
  // j>=8 -> K=16+(j-8); lanes 16-31 add +8 to K. (ISA 7.12.2)
  const _Float16* ap0 = Ag + (long)(r0 + lm) * Kd + lg * 8;
  // B layout (16-bit B 32x16): same K order as A: half j<8 -> K=8*(lane/16)+j,
  // j>=8 -> K=16+8*(lane/16)+(j-8); N = lane%16.
  const _Float16* bp0 = W + (long)(n0 + lm) * Kd + lg * 8;

  for (int kk = 0; kk < Kd; kk += 32) {
    v16h a, b;
    const _Float16* ap = ap0 + kk;
    const _Float16* bp = bp0 + kk;
    #pragma unroll
    for (int j = 0; j < 8; ++j) { a[j] = ap[j]; a[8 + j] = ap[16 + j]; }
    #pragma unroll
    for (int j = 0; j < 8; ++j) { b[j] = bp[j]; b[8 + j] = bp[16 + j]; }
    if (kk + 32 < Kd) __builtin_prefetch(ap + 32, 0, 1);   // global_prefetch_b8
    acc = __builtin_amdgcn_wmma_f32_16x16x32_f16(false, a, false, b,
                                                 (short)0, acc, false, false);
  }

  // C layout: lane -> N = n0 + lane%16 ; VGPR r -> M = r0 + r + 8*(lane/16)
  #pragma unroll
  for (int r = 0; r < 8; ++r) {
    int m = r0 + r + (lg << 3);
    float v = acc[r];
    if (act == 1)      v = fmaxf(v, 0.0f);
    else if (act == 2) v = tanhf(v);
    bool real = (m < Mg);
    long ci = (long)m * N + n0 + lm;
    if (Ch) Ch[(long)g * chStride + ci] = real ? (_Float16)v : (_Float16)0.0f;
    if (Co && real) Co[(long)g * coStride + ci] = v;
  }
}

// ---------------------------------------------------------------------------
// knn: one wave per 16-row block; Gram tile via 4 chained WMMAs (K=128),
// d2 = sq_i + sq_j - 2*dot, LDS-staged, per-row sorted top-8 insertion.
// ---------------------------------------------------------------------------
__global__ __launch_bounds__(32)
void knn_topk(const _Float16* __restrict__ Hh, const float* __restrict__ sq,
              const int* __restrict__ nCur,
              int* __restrict__ src, int* __restrict__ tgt)
{
  int g = blockIdx.y;
  int n = nCur[g];
  if (n <= 1) return;
  int r0 = blockIdx.x << 4;
  if (r0 >= n) return;
  int ke = dmin_(KNN_, n - 1);
  int l = threadIdx.x, lm = l & 15, lg = l >> 4;
  const _Float16* Hg  = Hh + (long)g * NP_ * HID_;
  const float*    sqg = sq + (long)g * NP_;

  v16h arow[4];
  {
    const _Float16* ap = Hg + (long)(r0 + lm) * HID_ + lg * 8;
    #pragma unroll
    for (int c = 0; c < 4; ++c) {
      #pragma unroll
      for (int j = 0; j < 8; ++j) {
        arow[c][j]     = ap[c * 32 + j];
        arow[c][8 + j] = ap[c * 32 + 16 + j];
      }
    }
  }

  float best[KNN_]; int bidx[KNN_];
  #pragma unroll
  for (int q = 0; q < KNN_; ++q) { best[q] = 3.0e38f; bidx[q] = 0; }

  __shared__ float tile[16][17];

  for (int c0 = 0; c0 < n; c0 += 16) {
    v8f acc;
    #pragma unroll
    for (int r = 0; r < 8; ++r) acc[r] = 0.0f;
    const _Float16* bp0 = Hg + (long)(c0 + lm) * HID_ + lg * 8;
    #pragma unroll
    for (int c = 0; c < 4; ++c) {
      v16h b;
      #pragma unroll
      for (int j = 0; j < 8; ++j) { b[j] = bp0[c * 32 + j]; b[8 + j] = bp0[c * 32 + 16 + j]; }
      acc = __builtin_amdgcn_wmma_f32_16x16x32_f16(false, arow[c], false, b,
                                                   (short)0, acc, false, false);
    }
    float csq = sqg[c0 + lm];
    #pragma unroll
    for (int r = 0; r < 8; ++r) {
      int m = r + (lg << 3);
      tile[m][lm] = sqg[r0 + m] + csq - 2.0f * acc[r];
    }
    __syncthreads();
    if (l < 16 && (r0 + l) < n) {
      int row = r0 + l;
      #pragma unroll
      for (int cc = 0; cc < 16; ++cc) {
        int col = c0 + cc;
        if (col < n && col != row) {
          float v = tile[l][cc];
          if (v < best[KNN_ - 1]) {
            best[KNN_ - 1] = v; bidx[KNN_ - 1] = col;
            #pragma unroll
            for (int q = KNN_ - 1; q > 0; --q) {
              if (best[q] < best[q - 1]) {
                float tb = best[q]; best[q] = best[q - 1]; best[q - 1] = tb;
                int   ti = bidx[q]; bidx[q] = bidx[q - 1]; bidx[q - 1] = ti;
              }
            }
          }
        }
      }
    }
    __syncthreads();
  }
  if (l < 16 && (r0 + l) < n) {
    int row = r0 + l;
    long eb = (long)g * EMAX_ + (long)row * ke;
    for (int t = 0; t < ke; ++t) { src[eb + t] = bidx[t]; tgt[eb + t] = row; }
  }
}

// ---------------------------------------------------------------------------
// Small VALU kernels
// ---------------------------------------------------------------------------
__global__ void f32_to_f16(const float* __restrict__ a, _Float16* __restrict__ o, int n)
{
  int i = blockIdx.x * blockDim.x + threadIdx.x;
  if (i < n) o[i] = (_Float16)a[i];
}

__global__ void input_layer1(const float* __restrict__ x, const float* __restrict__ dn,
                             const float* __restrict__ W, const float* __restrict__ b,
                             _Float16* __restrict__ H1)
{
  int t = blockIdx.x * blockDim.x + threadIdx.x;
  if (t >= (G_ * NP_) * 256) return;
  int row = t >> 8, o = t & 255;
  float s = b[o];
  #pragma unroll
  for (int c = 0; c < INDIM_; ++c) s += dn[c] * x[row * INDIM_ + c] * W[o * INDIM_ + c];
  H1[(long)row * 256 + o] = (_Float16)tanhf(s);
}

__global__ void init_graph(const float* __restrict__ Hf, float* __restrict__ sq,
                           int* __restrict__ nCur)
{
  int i = blockIdx.x * blockDim.x + threadIdx.x;
  if (i >= G_ * NP_) return;
  const float* h = Hf + (long)i * HID_;
  float s = 0.0f;
  for (int c = 0; c < HID_; ++c) s += h[c] * h[c];
  sq[i] = s;
  if ((i & (NP_ - 1)) == 0) nCur[i >> 11] = NP_;
}

// Build EdgeConv input rows: m_in[e] = [ h_tgt , h_src - h_tgt ]  (f16, padded)
__global__ void gather_edges(const float* __restrict__ Hf, const int* __restrict__ src,
                             const int* __restrict__ tgt, const int* __restrict__ nCur,
                             _Float16* __restrict__ mIn)
{
  int g = blockIdx.y;
  int n = nCur[g];
  if (n <= 1) return;
  int ke = dmin_(KNN_, n - 1);
  int E = n * ke, Ep = (E + 15) & ~15;
  long t = (long)blockIdx.x * blockDim.x + threadIdx.x;
  int e = (int)(t >> 8), c = (int)(t & 255);
  if (e >= Ep) return;
  _Float16* o = mIn + (long)g * EMAX_ * 256;
  if (e >= E) { o[(long)e * 256 + c] = (_Float16)0.0f; return; }
  int s  = src[(long)g * EMAX_ + e];
  int tt = tgt[(long)g * EMAX_ + e];
  const float* Hg = Hf + (long)g * NP_ * HID_;
  float v;
  if (c < HID_) v = Hg[(long)tt * HID_ + c];
  else { int cc = c - HID_; v = Hg[(long)s * HID_ + cc] - Hg[(long)tt * HID_ + cc]; }
  o[(long)e * 256 + c] = (_Float16)v;
}

// segment_sum over the k_eff consecutive edges of each target node
__global__ void reduce_hp(const float* __restrict__ m2, const int* __restrict__ nCur,
                          float* __restrict__ Hp)
{
  int g = blockIdx.y;
  int n = nCur[g];
  if (n <= 1) return;
  int t = blockIdx.x * blockDim.x + threadIdx.x;
  int i = t >> 7, c = t & 127;
  if (i >= n) return;
  int ke = dmin_(KNN_, n - 1);
  const float* mg = m2 + (long)g * EMAX_ * HID_ + (long)i * ke * HID_ + c;
  float s = 0.0f;
  for (int j = 0; j < ke; ++j) s += mg[(long)j * HID_];
  Hp[(long)g * NP_ * HID_ + (long)i * HID_ + c] = s;
}

__global__ void edge_score(const float* __restrict__ Hp, const int* __restrict__ src,
                           const int* __restrict__ tgt, const float* __restrict__ plW,
                           const float* __restrict__ plB, const int* __restrict__ nCur,
                           float* __restrict__ esc)
{
  int g = blockIdx.y;
  int n = nCur[g];
  if (n <= 1) return;
  int ke = dmin_(KNN_, n - 1), E = n * ke;
  int e = blockIdx.x * blockDim.x + threadIdx.x;
  if (e >= E) return;
  int s = src[(long)g * EMAX_ + e], t = tgt[(long)g * EMAX_ + e];
  const float* hs = Hp + (long)g * NP_ * HID_ + (long)s * HID_;
  const float* ht = Hp + (long)g * NP_ * HID_ + (long)t * HID_;
  float acc = plB[0];
  for (int c = 0; c < HID_; ++c) acc += plW[c] * hs[c] + plW[HID_ + c] * ht[c];
  esc[(long)g * EMAX_ + e] = acc;
}

__global__ void seg_softmax(const float* __restrict__ esc, const int* __restrict__ nCur,
                            float* __restrict__ scr)
{
  int g = blockIdx.y;
  int n = nCur[g];
  if (n <= 1) return;
  int i = blockIdx.x * blockDim.x + threadIdx.x;
  if (i >= n) return;
  int ke = dmin_(KNN_, n - 1);
  const float* eg = esc + (long)g * EMAX_ + (long)i * ke;
  float mx = -3.0e38f;
  for (int j = 0; j < ke; ++j) mx = fmaxf(mx, eg[j]);
  float den = 0.0f;
  for (int j = 0; j < ke; ++j) den += expf(eg[j] - mx);
  float* sg = scr + (long)g * EMAX_ + (long)i * ke;
  for (int j = 0; j < ke; ++j) sg[j] = expf(eg[j] - mx) / (den + 1e-16f) + 0.5f;
}

// descending-score sort == ascending sort on key = -score ; padding = +INF
__global__ void sort_init(const float* __restrict__ scr, const int* __restrict__ nCur,
                          float* __restrict__ key, int* __restrict__ val)
{
  int g = blockIdx.y;
  if (nCur[g] <= 1) return;
  int n = nCur[g];
  int ke = dmin_(KNN_, n - 1), E = n * ke;
  int i = blockIdx.x * blockDim.x + threadIdx.x;
  if (i >= SORTN_) return;
  key[(long)g * SORTN_ + i] = (i < E) ? -scr[(long)g * EMAX_ + i] : 3.0e38f;
  val[(long)g * SORTN_ + i] = i;
}

__global__ void bitonic_pass(float* __restrict__ key, int* __restrict__ val,
                             const int* __restrict__ nCur, int j, int k)
{
  int g = blockIdx.y;
  if (nCur[g] <= 1) return;
  int i = blockIdx.x * blockDim.x + threadIdx.x;
  int ixj = i ^ j;
  if (ixj <= i) return;
  long b = (long)g * SORTN_;
  float ki = key[b + i], kj = key[b + ixj];
  bool up = ((i & k) == 0);
  if ((ki > kj) == up) {
    key[b + i] = kj; key[b + ixj] = ki;
    int vi = val[b + i]; val[b + i] = val[b + ixj]; val[b + ixj] = vi;
  }
}

// Greedy maximal matching over edges in descending score order (serial scan,
// LDS-resident flags); assigns cluster ids exactly like the reference scan.
__global__ __launch_bounds__(256)
void greedy_merge(const int* __restrict__ nCur, const int* __restrict__ sval,
                  const float* __restrict__ scr, const int* __restrict__ src,
                  const int* __restrict__ tgt, int* __restrict__ cluster,
                  float* __restrict__ msc, int* __restrict__ nNew)
{
  __shared__ int rem[NP_];
  __shared__ int clus[NP_];
  int g = blockIdx.x;
  int n = nCur[g];
  if (n <= 1) return;
  for (int i = threadIdx.x; i < n; i += blockDim.x) { rem[i] = 1; clus[i] = 0; }
  __syncthreads();
  if (threadIdx.x == 0) {
    int ke = dmin_(KNN_, n - 1), E = n * ke;
    long sb = (long)g * SORTN_, eb = (long)g * EMAX_;
    int cnt = 0;
    for (int r = 0; r < E; ++r) {
      int e = sval[sb + r];
      int s = src[eb + e], t = tgt[eb + e];
      if (rem[s] && rem[t]) {
        clus[s] = cnt; clus[t] = cnt; rem[s] = 0; rem[t] = 0;
        msc[(long)g * NP_ + cnt] = scr[eb + e];
        ++cnt;
      }
    }
    int nm = cnt;
    for (int i = 0; i < n; ++i)
      if (rem[i]) { clus[i] = nm; msc[(long)g * NP_ + nm] = 1.0f; ++nm; }
    nNew[g] = nm;
  }
  __syncthreads();
  for (int i = threadIdx.x; i < n; i += blockDim.x)
    cluster[(long)g * NP_ + i] = clus[i];
}

__global__ void zero_f32(float* __restrict__ p, long n)
{
  long i = (long)blockIdx.x * blockDim.x + threadIdx.x;
  if (i < n) p[i] = 0.0f;
}

__global__ void scatter_pool(const float* __restrict__ Hp, const int* __restrict__ cluster,
                             const int* __restrict__ nCur, float* __restrict__ newh)
{
  int g = blockIdx.y;
  int n = nCur[g];
  if (n <= 1) return;
  int t = blockIdx.x * blockDim.x + threadIdx.x;
  int i = t >> 7, c = t & 127;
  if (i >= n) return;
  atomicAdd(&newh[(long)g * NP_ * HID_ + (long)cluster[(long)g * NP_ + i] * HID_ + c],
            Hp[(long)g * NP_ * HID_ + (long)i * HID_ + c]);
}

__global__ void finalize_level(const float* __restrict__ newh, const float* __restrict__ msc,
                               const int* __restrict__ nNew, const int* __restrict__ nCur,
                               float* __restrict__ Hf, _Float16* __restrict__ Hh,
                               float* __restrict__ sq)
{
  int g = blockIdx.y;
  if (nCur[g] <= 1) return;
  int i = blockIdx.x * blockDim.x + threadIdx.x;
  int nm = nNew[g];
  if (i >= nm) return;
  float m = msc[(long)g * NP_ + i];
  const float* sp = newh + (long)g * NP_ * HID_ + (long)i * HID_;
  float*    df = Hf + (long)g * NP_ * HID_ + (long)i * HID_;
  _Float16* dh = Hh + (long)g * NP_ * HID_ + (long)i * HID_;
  float s = 0.0f;
  for (int c = 0; c < HID_; ++c) {
    float v = sp[c] * m;
    df[c] = v; dh[c] = (_Float16)v; s += v * v;
  }
  sq[(long)g * NP_ + i] = s;
}

__global__ void advance_n(int* __restrict__ nCur, const int* __restrict__ nNew)
{
  int g = threadIdx.x;
  if (g < G_ && nCur[g] > 1) nCur[g] = nNew[g];
}

__global__ __launch_bounds__(256)
void out_mlp(const float* __restrict__ Hf,
             const float* __restrict__ W1, const float* __restrict__ b1,
             const float* __restrict__ W2, const float* __restrict__ b2,
             const float* __restrict__ W3, const float* __restrict__ b3,
             const float* __restrict__ W4, const float* __restrict__ b4,
             const float* __restrict__ W5, const float* __restrict__ b5,
             float* __restrict__ out)
{
  __shared__ float h0[HID_], y1[256], y2[256], y3[HID_], y4[64];
  int g = blockIdx.x, t = threadIdx.x;
  if (t < HID_) h0[t] = Hf[(long)g * NP_ * HID_ + t];
  __syncthreads();
  { float s = b1[t]; for (int c = 0; c < HID_; ++c) s += W1[t * HID_ + c] * h0[c];
    y1[t] = fmaxf(s, 0.0f); }
  __syncthreads();
  { float s = b2[t]; for (int c = 0; c < 256; ++c) s += W2[t * 256 + c] * y1[c];
    y2[t] = fmaxf(s, 0.0f); }
  __syncthreads();
  if (t < HID_) { float s = b3[t]; for (int c = 0; c < 256; ++c) s += W3[t * 256 + c] * y2[c];
    y3[t] = fmaxf(s, 0.0f); }
  __syncthreads();
  if (t < 64) { float s = b4[t]; for (int c = 0; c < HID_; ++c) s += W4[t * HID_ + c] * y3[c];
    y4[t] = fmaxf(s, 0.0f); }
  __syncthreads();
  if (t == 0) { float s = b5[0]; for (int c = 0; c < 64; ++c) s += W5[c] * y4[c];
    out[g] = s; }
}

// ---------------------------------------------------------------------------
// Host side
// ---------------------------------------------------------------------------
extern "C" void kernel_launch(void* const* d_in, const int* in_sizes, int n_in,
                              void* d_out, int out_size, void* d_ws, size_t ws_size,
                              hipStream_t stream)
{
  const float* x      = (const float*)d_in[0];
  // d_in[1] = batch (structure is known: g = i / NP_), unused
  const float* dn     = (const float*)d_in[2];
  const float* in_W1  = (const float*)d_in[3];  const float* in_b1 = (const float*)d_in[4];
  const float* in_W2  = (const float*)d_in[5];  const float* in_b2 = (const float*)d_in[6];
  const float* in_W3  = (const float*)d_in[7];  const float* in_b3 = (const float*)d_in[8];
  const float* cv_W1  = (const float*)d_in[9];  const float* cv_b1 = (const float*)d_in[10];
  const float* cv_W2  = (const float*)d_in[11]; const float* cv_b2 = (const float*)d_in[12];
  const float* pl_W   = (const float*)d_in[13]; const float* pl_b  = (const float*)d_in[14];
  const float* o_W1   = (const float*)d_in[15]; const float* o_b1  = (const float*)d_in[16];
  const float* o_W2   = (const float*)d_in[17]; const float* o_b2  = (const float*)d_in[18];
  const float* o_W3   = (const float*)d_in[19]; const float* o_b3  = (const float*)d_in[20];
  const float* o_W4   = (const float*)d_in[21]; const float* o_b4  = (const float*)d_in[22];
  const float* o_W5   = (const float*)d_in[23]; const float* o_b5  = (const float*)d_in[24];
  float* out = (float*)d_out;

  char* p = (char*)d_ws;
  auto alloc = [&](size_t bytes) -> void* {
    char* r = p;
    p += (bytes + 255) & ~(size_t)255;
    return (void*)r;
  };

  _Float16* H1h   = (_Float16*)alloc((size_t)G_*NP_*256*2);
  _Float16* H2h   = (_Float16*)alloc((size_t)G_*NP_*256*2);
  float*    Hf32  = (float*)   alloc((size_t)G_*NP_*HID_*4);
  _Float16* Hf16  = (_Float16*)alloc((size_t)G_*NP_*HID_*2);
  float*    sq    = (float*)   alloc((size_t)G_*NP_*4);
  _Float16* W2h   = (_Float16*)alloc((size_t)256*256*2);
  _Float16* W3h   = (_Float16*)alloc((size_t)128*256*2);
  _Float16* cW1h  = (_Float16*)alloc((size_t)192*256*2);
  _Float16* cW2h  = (_Float16*)alloc((size_t)128*192*2);
  int*      srcB  = (int*)     alloc((size_t)G_*EMAX_*4);
  int*      tgtB  = (int*)     alloc((size_t)G_*EMAX_*4);
  _Float16* mIn   = (_Float16*)alloc((size_t)G_*EMAX_*256*2);
  _Float16* m1h   = (_Float16*)alloc((size_t)G_*EMAX_*192*2);
  float*    m2f   = (float*)   alloc((size_t)G_*EMAX_*128*4);
  float*    Hp    = (float*)   alloc((size_t)G_*NP_*HID_*4);
  float*    esc   = (float*)   alloc((size_t)G_*EMAX_*4);
  float*    scr   = (float*)   alloc((size_t)G_*EMAX_*4);
  float*    skey  = (float*)   alloc((size_t)G_*SORTN_*4);
  int*      sval  = (int*)     alloc((size_t)G_*SORTN_*4);
  int*      clus  = (int*)     alloc((size_t)G_*NP_*4);
  float*    msc   = (float*)   alloc((size_t)G_*NP_*4);
  float*    newh  = (float*)   alloc((size_t)G_*NP_*HID_*4);
  int*      nCur  = (int*)     alloc((size_t)G_*4);
  int*      nNew  = (int*)     alloc((size_t)G_*4);

  if ((size_t)(p - (char*)d_ws) > ws_size) return;  // workspace too small

  // ---- weight conversion to f16 ----
  f32_to_f16<<<(256*256+255)/256, 256, 0, stream>>>(in_W2, W2h, 256*256);
  f32_to_f16<<<(128*256+255)/256, 256, 0, stream>>>(in_W3, W3h, 128*256);
  f32_to_f16<<<(192*256+255)/256, 256, 0, stream>>>(cv_W1, cW1h, 192*256);
  f32_to_f16<<<(128*192+255)/256, 256, 0, stream>>>(cv_W2, cW2h, 128*192);

  // ---- input MLP ----
  input_layer1<<<(G_*NP_*256)/256, 256, 0, stream>>>(x, dn, in_W1, in_b1, H1h);
  gemm_wmma<<<dim3(256/16, (G_*NP_)/16, 1), 32, 0, stream>>>(
      H1h, 0, W2h, in_b2, nullptr, 0, H2h, 0, nCur, 0, G_*NP_, 256, 256, 2);
  gemm_wmma<<<dim3(128/16, (G_*NP_)/16, 1), 32, 0, stream>>>(
      H2h, 0, W3h, in_b3, Hf32, 0, Hf16, 0, nCur, 0, G_*NP_, 128, 256, 2);
  init_graph<<<(G_*NP_)/256, 256, 0, stream>>>(Hf32, sq, nCur);

  // ---- dynamic reduction levels ----
  for (int lvl = 0; lvl < MAXLVL_; ++lvl) {
    knn_topk<<<dim3(NP_/16, G_), 32, 0, stream>>>(Hf16, sq, nCur, srcB, tgtB);
    gather_edges<<<dim3(EMAX_, G_), 256, 0, stream>>>(Hf32, srcB, tgtB, nCur, mIn);
    gemm_wmma<<<dim3(192/16, EMAX_/16, G_), 32, 0, stream>>>(
        mIn, (long)EMAX_*256, cW1h, cv_b1, nullptr, 0, m1h, (long)EMAX_*192,
        nCur, 1, 0, 192, 256, 1);
    gemm_wmma<<<dim3(128/16, EMAX_/16, G_), 32, 0, stream>>>(
        m1h, (long)EMAX_*192, cW2h, cv_b2, m2f, (long)EMAX_*128, nullptr, 0,
        nCur, 1, 0, 128, 192, 1);
    reduce_hp<<<dim3((NP_*HID_)/256, G_), 256, 0, stream>>>(m2f, nCur, Hp);
    edge_score<<<dim3(EMAX_/256, G_), 256, 0, stream>>>(Hp, srcB, tgtB, pl_W, pl_b,
                                                        nCur, esc);
    seg_softmax<<<dim3(NP_/256, G_), 256, 0, stream>>>(esc, nCur, scr);
    sort_init<<<dim3(SORTN_/256, G_), 256, 0, stream>>>(scr, nCur, skey, sval);
    for (int k = 2; k <= SORTN_; k <<= 1)
      for (int j = k >> 1; j > 0; j >>= 1)
        bitonic_pass<<<dim3(SORTN_/256, G_), 256, 0, stream>>>(skey, sval, nCur, j, k);
    greedy_merge<<<G_, 256, 0, stream>>>(nCur, sval, scr, srcB, tgtB, clus, msc, nNew);
    zero_f32<<<(G_*NP_*HID_)/256, 256, 0, stream>>>(newh, (long)G_*NP_*HID_);
    scatter_pool<<<dim3((NP_*HID_)/256, G_), 256, 0, stream>>>(Hp, clus, nCur, newh);
    finalize_level<<<dim3(NP_/256, G_), 256, 0, stream>>>(newh, msc, nNew, nCur,
                                                          Hf32, Hf16, sq);
    advance_n<<<1, G_, 0, stream>>>(nCur, nNew);
  }

  // ---- output MLP ----
  out_mlp<<<G_, 256, 0, stream>>>(Hf32, o_W1, o_b1, o_W2, o_b2, o_W3, o_b3,
                                  o_W4, o_b4, o_W5, o_b5, out);
}
